// AdvancedGNNPredictor_53420803228012
// MI455X (gfx1250) — matrix operand order repaired and checked by reading, not compiled, not run on hardware
//
#include <hip/hip_runtime.h>
#include <math.h>

// ---------------------------------------------------------------------------
// Types for CDNA5 WMMA
// ---------------------------------------------------------------------------
typedef __attribute__((ext_vector_type(2)))  float  v2f;
typedef __attribute__((ext_vector_type(8)))  float  v8f;
typedef __attribute__((ext_vector_type(16))) __bf16 v16bf;

#define NEG_SLOPE 0.2f
#define EPS_BN    1e-5f
#define N_GRAPHS  64

// ---------------------------------------------------------------------------
// float atomic max via CAS (portable, monotone; NaN-free inputs here)
// ---------------------------------------------------------------------------
__device__ __forceinline__ void atomicMaxF32(float* addr, float val) {
  unsigned int* ua = reinterpret_cast<unsigned int*>(addr);
  unsigned int cur = *ua;
  while (__uint_as_float(cur) < val) {
    unsigned int assumed = cur;
    cur = atomicCAS(ua, assumed, __float_as_uint(val));
    if (cur == assumed) break;
  }
}

// ---------------------------------------------------------------------------
// WMMA GEMM: C[M,N] = A[M,K] @ B[K,N], all row-major fp32.
// One wave32 computes one 16x16 tile. M,N multiples of 16; K multiple of 4
// (and of 32 for the bf16 fallback). Wave-uniform guard keeps EXEC all-ones
// around the WMMA ops.
// ---------------------------------------------------------------------------
#if __has_builtin(__builtin_amdgcn_wmma_f32_16x16x4_f32)

__global__ void gemm_wmma(const float* __restrict__ A, const float* __restrict__ B,
                          float* __restrict__ C, int M, int K, int N, int tiles) {
  int wid  = (int)((blockIdx.x * (unsigned)blockDim.x + threadIdx.x) >> 5);
  if (wid >= tiles) return;                 // whole-wave uniform exit
  int lane = threadIdx.x & 31;
  int tilesN = N >> 4;
  int tm = wid / tilesN, tn = wid % tilesN;
  int lh = lane >> 4;                       // lane-half: 0 or 1
  int li = lane & 15;
  size_t row = (size_t)(tm * 16 + li);      // A row handled by this lane
  int    col = tn * 16 + li;                // B/C column handled by this lane
  v8f acc = {};
  for (int k0 = 0; k0 < K; k0 += 4) {
    int ka = k0 + 2 * lh;                   // lanes 0-15: K=k0,k0+1 ; lanes 16-31: K=k0+2,k0+3
    v2f av, bv;
    av.x = A[row * (size_t)K + ka];
    av.y = A[row * (size_t)K + ka + 1];
    bv.x = B[(size_t)ka * N + col];
    bv.y = B[(size_t)(ka + 1) * N + col];
    acc = __builtin_amdgcn_wmma_f32_16x16x4_f32(false, av, false, bv,
                                                (short)0, acc, false, false);
  }
  // D layout: VGPR r -> M = 16*tm + r + 8*lh, N = col
  size_t orow = (size_t)(tm * 16 + 8 * lh);
  #pragma unroll
  for (int r = 0; r < 8; ++r)
    C[(orow + r) * (size_t)N + col] = acc[r];
}

#else  // fallback: bf16 WMMA (codegen-confirmed builtin), on-the-fly conversion

__global__ void gemm_wmma(const float* __restrict__ A, const float* __restrict__ B,
                          float* __restrict__ C, int M, int K, int N, int tiles) {
  int wid  = (int)((blockIdx.x * (unsigned)blockDim.x + threadIdx.x) >> 5);
  if (wid >= tiles) return;
  int lane = threadIdx.x & 31;
  int tilesN = N >> 4;
  int tm = wid / tilesN, tn = wid % tilesN;
  int lh = lane >> 4;
  int li = lane & 15;
  size_t row = (size_t)(tm * 16 + li);
  int    col = tn * 16 + li;
  v8f acc = {};
  for (int k0 = 0; k0 < K; k0 += 32) {
    v16bf av, bv;
    #pragma unroll
    for (int j = 0; j < 8; ++j) {
      // A 16x32 bf16 layout (ISA table): lanes0-15 VGPR0..3 K=0..7, VGPR4..7 K=16..23;
      // lanes16-31 shifted by 8.
      int ka = k0 + 2 * j + ((j >= 4) ? 8 : 0) + 8 * lh;
      av[2 * j]     = (__bf16)A[row * (size_t)K + ka];
      av[2 * j + 1] = (__bf16)A[row * (size_t)K + ka + 1];
      // B 32x16 bf16 layout: same K order as A per slot, lane -> N
      int kb = ka;
      int cb = col;
      bv[2 * j]     = (__bf16)B[(size_t)kb * N + cb];
      bv[2 * j + 1] = (__bf16)B[(size_t)(kb + 1) * N + cb];
    }
    acc = __builtin_amdgcn_wmma_f32_16x16x32_bf16(false, av, false, bv,
                                                  (short)0, acc, false, false);
  }
  size_t orow = (size_t)(tm * 16 + 8 * lh);
  #pragma unroll
  for (int r = 0; r < 8; ++r)
    C[(orow + r) * (size_t)N + col] = acc[r];
}

#endif

// ---------------------------------------------------------------------------
// Utility: fill float buffer with constant
// ---------------------------------------------------------------------------
__global__ void fill_f32(float* __restrict__ p, float v, size_t n) {
  size_t i = (size_t)blockIdx.x * blockDim.x + threadIdx.x;
  if (i < n) p[i] = v;
}

// ---------------------------------------------------------------------------
// Per-node attention scores: aS[n,h] = <H[n,h,:], att_s[h,:]>, same for aD.
// ---------------------------------------------------------------------------
__global__ void node_scores(const float* __restrict__ H,
                            const float* __restrict__ att_s,
                            const float* __restrict__ att_d,
                            float* __restrict__ aS, float* __restrict__ aD,
                            int Nn, int Hh, int C) {
  int t = blockIdx.x * blockDim.x + threadIdx.x;
  if (t >= Nn * Hh) return;
  int n = t / Hh, h = t % Hh;
  const float* hp  = H + ((size_t)n * Hh + h) * C;
  const float* asp = att_s + (size_t)h * C;
  const float* adp = att_d + (size_t)h * C;
  float s = 0.f, d = 0.f;
  for (int c = 0; c < C; ++c) { float v = hp[c]; s += v * asp[c]; d += v * adp[c]; }
  aS[t] = s;
  aD[t] = d;
}

// ---------------------------------------------------------------------------
// Edge pass 1: e = leaky_relu(aS[src]+aD[dst]); store; segment-max into mbuf[dst].
// Edges e >= E are the self loops (src=dst=e-E).
// ---------------------------------------------------------------------------
__global__ void edge_scores_max(const int* __restrict__ src, const int* __restrict__ dst,
                                int E, int Nn, int Hh,
                                const float* __restrict__ aS, const float* __restrict__ aD,
                                float* __restrict__ esc, float* __restrict__ mbuf) {
  long long t = (long long)blockIdx.x * blockDim.x + threadIdx.x;
  long long total = (long long)(E + Nn) * Hh;
  if (t >= total) return;
  int e = (int)(t / Hh), h = (int)(t % Hh);
  int s = (e < E) ? src[e] : (e - E);
  int d = (e < E) ? dst[e] : (e - E);
  float x = aS[(size_t)s * Hh + h] + aD[(size_t)d * Hh + h];
  x = (x > 0.f) ? x : NEG_SLOPE * x;
  esc[t] = x;
  atomicMaxF32(&mbuf[(size_t)d * Hh + h], x);
}

// ---------------------------------------------------------------------------
// Edge pass 2: den[dst] += exp(e - m[dst])
// ---------------------------------------------------------------------------
__global__ void edge_expsum(const int* __restrict__ dst, int E, int Nn, int Hh,
                            const float* __restrict__ esc, const float* __restrict__ mbuf,
                            float* __restrict__ den) {
  long long t = (long long)blockIdx.x * blockDim.x + threadIdx.x;
  long long total = (long long)(E + Nn) * Hh;
  if (t >= total) return;
  int e = (int)(t / Hh), h = (int)(t % Hh);
  int d = (e < E) ? dst[e] : (e - E);
  float ex = __expf(esc[t] - mbuf[(size_t)d * Hh + h]);
  atomicAdd(&den[(size_t)d * Hh + h], ex);
}

// ---------------------------------------------------------------------------
// Edge pass 3: out[dst,h,c] += alpha * H[src,h,c]
// ---------------------------------------------------------------------------
__global__ void edge_scatter(const int* __restrict__ src, const int* __restrict__ dst,
                             int E, int Nn, int Hh, int C,
                             const float* __restrict__ H, const float* __restrict__ esc,
                             const float* __restrict__ mbuf, const float* __restrict__ den,
                             float* __restrict__ out) {
  long long t = (long long)blockIdx.x * blockDim.x + threadIdx.x;
  long long total = (long long)(E + Nn) * Hh * C;
  if (t >= total) return;
  int hc = (int)(t % (Hh * C));
  long long e = t / (Hh * C);
  int h = hc / C, c = hc % C;
  int s = (e < E) ? src[e] : (int)(e - E);
  int d = (e < E) ? dst[e] : (int)(e - E);
  float ex    = __expf(esc[e * Hh + h] - mbuf[(size_t)d * Hh + h]);
  float alpha = ex / (den[(size_t)d * Hh + h] + 1e-16f);
  atomicAdd(&out[((size_t)d * Hh + h) * C + c],
            H[((size_t)s * Hh + h) * C + c] * alpha);
}

// ---------------------------------------------------------------------------
// BN stats: per-channel sum / sumsq of (X + bias). blockDim.x == C.
// ---------------------------------------------------------------------------
#define BN_ROWS_PER_BLOCK 256
__global__ void bn_stats(const float* __restrict__ X, const float* __restrict__ bias,
                         int Nn, int C, float* __restrict__ bsum, float* __restrict__ bsq) {
  int c = threadIdx.x;
  size_t r0 = (size_t)blockIdx.x * BN_ROWS_PER_BLOCK;
  size_t r1 = r0 + BN_ROWS_PER_BLOCK; if (r1 > (size_t)Nn) r1 = (size_t)Nn;
  float b = bias[c];
  float s = 0.f, q = 0.f;
  for (size_t r = r0; r < r1; ++r) {
    float v = X[r * C + c] + b;
    s += v; q += v * v;
  }
  atomicAdd(&bsum[c], s);
  atomicAdd(&bsq[c], q);
}

// ---------------------------------------------------------------------------
// Apply bias + BN + ELU in place.
// ---------------------------------------------------------------------------
__global__ void bn_apply_elu(float* __restrict__ X, const float* __restrict__ bias,
                             const float* __restrict__ g, const float* __restrict__ bt,
                             const float* __restrict__ bsum, const float* __restrict__ bsq,
                             int Nn, int C) {
  size_t idx = (size_t)blockIdx.x * blockDim.x + threadIdx.x;
  size_t total = (size_t)Nn * C;
  if (idx >= total) return;
  int c = (int)(idx % C);
  float invn = 1.0f / (float)Nn;
  float mu   = bsum[c] * invn;
  float var  = bsq[c] * invn - mu * mu;
  float v = (X[idx] + bias[c] - mu) * rsqrtf(var + EPS_BN) * g[c] + bt[c];
  X[idx] = (v > 0.f) ? v : (__expf(v) - 1.0f);
}

// ---------------------------------------------------------------------------
// Global mean pool (accumulation part)
// ---------------------------------------------------------------------------
__global__ void pool_accum(const float* __restrict__ H3, const int* __restrict__ batch,
                           float* __restrict__ psum, float* __restrict__ pcnt,
                           int Nn, int C) {
  size_t idx = (size_t)blockIdx.x * blockDim.x + threadIdx.x;
  size_t total = (size_t)Nn * C;
  if (idx >= total) return;
  int n = (int)(idx / C), c = (int)(idx % C);
  int gidx = batch[n];
  atomicAdd(&psum[(size_t)gidx * C + c], H3[idx]);
  if (c == 0) atomicAdd(&pcnt[gidx], 1.0f);
}

// ---------------------------------------------------------------------------
// MLP head: 64 graphs x (64 -> 128 -> 32 -> 2), one block, staged in LDS.
// ---------------------------------------------------------------------------
__global__ void mlp_head(const float* __restrict__ psum, const float* __restrict__ pcnt,
                         const float* __restrict__ fw1, const float* __restrict__ fb1,
                         const float* __restrict__ fw2, const float* __restrict__ fb2,
                         const float* __restrict__ fw3, const float* __restrict__ fb3,
                         float* __restrict__ out) {
  __shared__ float pooled[N_GRAPHS * 64];
  __shared__ float z1[N_GRAPHS * 128];
  __shared__ float z2[N_GRAPHS * 32];
  int t = threadIdx.x, nt = blockDim.x;
  for (int i = t; i < N_GRAPHS * 64; i += nt) {
    int g = i / 64;
    pooled[i] = psum[i] / fmaxf(pcnt[g], 1.0f);
  }
  __syncthreads();
  for (int i = t; i < N_GRAPHS * 128; i += nt) {
    int g = i / 128, j = i % 128;
    float acc = fb1[j];
    for (int k = 0; k < 64; ++k) acc += pooled[g * 64 + k] * fw1[k * 128 + j];
    z1[i] = (acc > 0.f) ? acc : NEG_SLOPE * acc;
  }
  __syncthreads();
  for (int i = t; i < N_GRAPHS * 32; i += nt) {
    int g = i / 32, j = i % 32;
    float acc = fb2[j];
    for (int k = 0; k < 128; ++k) acc += z1[g * 128 + k] * fw2[k * 32 + j];
    z2[i] = (acc > 0.f) ? acc : NEG_SLOPE * acc;
  }
  __syncthreads();
  for (int i = t; i < N_GRAPHS * 2; i += nt) {
    int g = i / 2, j = i % 2;
    float acc = fb3[j];
    for (int k = 0; k < 32; ++k) acc += z2[g * 32 + k] * fw3[k * 2 + j];
    out[i] = acc;
  }
}

// ===========================================================================
// Host-side orchestration
// ===========================================================================
static inline void fillf(float* p, float v, size_t n, hipStream_t s) {
  unsigned blocks = (unsigned)((n + 255) / 256);
  fill_f32<<<blocks, 256, 0, s>>>(p, v, n);
}

static void run_gemm(const float* A, const float* B, float* C,
                     int M, int K, int N, hipStream_t s) {
  int tiles = (M / 16) * (N / 16);
  int wavesPerBlock = 4;                       // 128 threads
  unsigned blocks = (unsigned)((tiles + wavesPerBlock - 1) / wavesPerBlock);
  gemm_wmma<<<blocks, 32 * wavesPerBlock, 0, s>>>(A, B, C, M, K, N, tiles);
}

// One full GAT layer: H = Xin@W (via WMMA) -> attention -> scatter into Obuf,
// then bias+BN+ELU in place on Obuf.
static void run_gat_layer(const float* Xin, const float* W,
                          const float* att_s, const float* att_d, const float* bias,
                          const float* g, const float* bt,
                          const int* src, const int* dst, int E, int Nn,
                          int Fin, int Hh, int C,
                          float* Hbuf, float* Obuf,
                          float* aS, float* aD, float* mbuf, float* den, float* esc,
                          float* bsum, float* bsq, hipStream_t s) {
  int Fout = Hh * C;
  run_gemm(Xin, W, Hbuf, Nn, Fin, Fout, s);

  {
    int total = Nn * Hh;
    node_scores<<<(unsigned)((total + 255) / 256), 256, 0, s>>>(
        Hbuf, att_s, att_d, aS, aD, Nn, Hh, C);
  }

  fillf(mbuf, -3.0e38f, (size_t)Nn * Hh, s);
  fillf(den, 0.0f, (size_t)Nn * Hh, s);
  fillf(Obuf, 0.0f, (size_t)Nn * Fout, s);

  long long Et = (long long)E + Nn;
  {
    long long total = Et * Hh;
    edge_scores_max<<<(unsigned)((total + 255) / 256), 256, 0, s>>>(
        src, dst, E, Nn, Hh, aS, aD, esc, mbuf);
    edge_expsum<<<(unsigned)((total + 255) / 256), 256, 0, s>>>(
        dst, E, Nn, Hh, esc, mbuf, den);
  }
  {
    long long total = Et * Hh * C;
    edge_scatter<<<(unsigned)((total + 255) / 256), 256, 0, s>>>(
        src, dst, E, Nn, Hh, C, Hbuf, esc, mbuf, den, Obuf);
  }

  fillf(bsum, 0.0f, (size_t)Fout, s);
  fillf(bsq, 0.0f, (size_t)Fout, s);
  {
    unsigned blocks = (unsigned)((Nn + BN_ROWS_PER_BLOCK - 1) / BN_ROWS_PER_BLOCK);
    bn_stats<<<blocks, Fout, 0, s>>>(Obuf, bias, Nn, Fout, bsum, bsq);
  }
  {
    size_t total = (size_t)Nn * Fout;
    bn_apply_elu<<<(unsigned)((total + 255) / 256), 256, 0, s>>>(
        Obuf, bias, g, bt, bsum, bsq, Nn, Fout);
  }
}

extern "C" void kernel_launch(void* const* d_in, const int* in_sizes, int n_in,
                              void* d_out, int out_size, void* d_ws, size_t ws_size,
                              hipStream_t stream) {
  // Input order follows setup_inputs():
  const float* x    = (const float*)d_in[0];
  const int*   ei   = (const int*)  d_in[1];
  const int*   batch= (const int*)  d_in[2];
  const float* W1   = (const float*)d_in[3];
  const float* as1  = (const float*)d_in[4];
  const float* ad1  = (const float*)d_in[5];
  const float* b1   = (const float*)d_in[6];
  const float* W2   = (const float*)d_in[7];
  const float* as2  = (const float*)d_in[8];
  const float* ad2  = (const float*)d_in[9];
  const float* b2   = (const float*)d_in[10];
  const float* W3   = (const float*)d_in[11];
  const float* as3  = (const float*)d_in[12];
  const float* ad3  = (const float*)d_in[13];
  const float* b3   = (const float*)d_in[14];
  const float* g1   = (const float*)d_in[15];
  const float* bt1  = (const float*)d_in[16];
  const float* g2   = (const float*)d_in[17];
  const float* bt2  = (const float*)d_in[18];
  const float* g3   = (const float*)d_in[19];
  const float* bt3  = (const float*)d_in[20];
  const float* fw1  = (const float*)d_in[21];
  const float* fb1  = (const float*)d_in[22];
  const float* fw2  = (const float*)d_in[23];
  const float* fb2  = (const float*)d_in[24];
  const float* fw3  = (const float*)d_in[25];
  const float* fb3  = (const float*)d_in[26];

  const int Nn = in_sizes[0] / 128;     // 50000 (multiple of 16)
  const int E  = in_sizes[1] / 2;       // 800000
  const int* src = ei;
  const int* dst = ei + E;

  // ---- workspace layout (floats) ----
  float* bufA = (float*)d_ws;                      // [Nn*256] GEMM output H
  float* bufB = bufA + (size_t)Nn * 256;           // [Nn*256] aggregation / next input
  float* aS   = bufB + (size_t)Nn * 256;           // [Nn*4]
  float* aD   = aS   + (size_t)Nn * 4;             // [Nn*4]
  float* mbuf = aD   + (size_t)Nn * 4;             // [Nn*4]
  float* den  = mbuf + (size_t)Nn * 4;             // [Nn*4]
  float* esc  = den  + (size_t)Nn * 4;             // [(E+Nn)*4]
  float* bsum = esc  + (size_t)(E + Nn) * 4;       // [256]
  float* bsq  = bsum + 256;                        // [256]
  float* psum = bsq  + 256;                        // [64*64]
  float* pcnt = psum + (size_t)N_GRAPHS * 64;      // [64]
  size_t needF = (size_t)(pcnt + N_GRAPHS - (float*)d_ws);
  if (ws_size < needF * sizeof(float)) return;     // workspace too small: no-op

  // ---- Layer 1: 128 -> 4 heads x 64, concat ----
  run_gat_layer(x, W1, as1, ad1, b1, g1, bt1, src, dst, E, Nn,
                128, 4, 64, bufA, bufB, aS, aD, mbuf, den, esc, bsum, bsq, stream);

  // ---- Layer 2: 256 -> 4 heads x 32, concat ----
  run_gat_layer(bufB, W2, as2, ad2, b2, g2, bt2, src, dst, E, Nn,
                256, 4, 32, bufA, bufB, aS, aD, mbuf, den, esc, bsum, bsq, stream);

  // ---- Layer 3: 128 -> 1 head x 64, mean(=identity for 1 head) ----
  run_gat_layer(bufB, W3, as3, ad3, b3, g3, bt3, src, dst, E, Nn,
                128, 1, 64, bufA, bufB, aS, aD, mbuf, den, esc, bsum, bsq, stream);

  // ---- Global mean pool + MLP head ----
  fillf(psum, 0.0f, (size_t)N_GRAPHS * 64, stream);
  fillf(pcnt, 0.0f, (size_t)N_GRAPHS, stream);
  {
    size_t total = (size_t)Nn * 64;
    pool_accum<<<(unsigned)((total + 255) / 256), 256, 0, stream>>>(
        bufB, batch, psum, pcnt, Nn, 64);
  }
  mlp_head<<<1, 256, 0, stream>>>(psum, pcnt, fw1, fb1, fw2, fb2, fw3, fb3,
                                  (float*)d_out);
}
